// VideoResNet_15281493639879
// MI455X (gfx1250) — matrix operand order repaired and checked by reading, not compiled, not run on hardware
//
#include <hip/hip_runtime.h>

// ---------------------------------------------------------------------------
// VideoResNet forward for MI455X (gfx1250, wave32).
// Convs = implicit GEMM on v_wmma_f32_16x16x32_bf16.
// Activations: channels-last bf16. Weights: device-prepacked into the native
// WMMA B-fragment layout so the GEMM loop is pure b128 loads + WMMA.
// Loop structure: tap-outer (mixed-radix increment, no div/mod in kernel
// body), channel-inner (Cin % 32 == 0 for every conv in this net).
// Padding taps are handled by redirecting the A pointer at a zero-filled
// pad page, so the inner loop has no predication at all.
// ---------------------------------------------------------------------------

typedef unsigned short u16t;
typedef __attribute__((ext_vector_type(16))) __bf16       v16bf;
typedef __attribute__((ext_vector_type(8)))  float        v8f;
typedef __attribute__((ext_vector_type(4)))  unsigned int v4u;

union ABFrag { v4u q[2]; v16bf v; };

__device__ __forceinline__ u16t f2bf(float f) {
  unsigned int u = __builtin_bit_cast(unsigned int, f);
  u += 0x7FFFu + ((u >> 16) & 1u);   // round-to-nearest-even
  return (u16t)(u >> 16);
}
__device__ __forceinline__ float bf2f(u16t h) {
  unsigned int u = ((unsigned int)h) << 16;
  return __builtin_bit_cast(float, u);
}

struct ConvDesc {
  int nd, N;
  int in_dim[4], out_dim[4], ks[4], stride[4], pad[4];
  int Cin, Cout, K, M, taps;
};

#define ZPAD_ELEMS 4096   // 8KB zero page >= max Cin (1024) + run offsets

__global__ void zero_fill_kernel(u16t* __restrict__ p, int n) {
  int i = blockIdx.x * blockDim.x + threadIdx.x;
  if (i < n) p[i] = 0;
}

// ---------------------------------------------------------------------------
// Implicit-GEMM conv. Wave tile: 16(M) x 64(N); block = 8 waves = 128 M rows.
// A fragment (16-bit A 16x32 ISA layout): lane holds row M=lane&15; its 16
// bf16 values are two contiguous K-runs of 8:  c = cb + half*8 + {0..7} and
// c = cb + 16 + half*8 + {0..7}. Channels-last + Cin % 32 == 0 makes each
// run one aligned, UNCONDITIONAL 16B load: out-of-bounds taps redirect the
// per-tap base pointer to a zero page instead of predicating the loads.
// B fragments are pre-packed; per K-step they are 8 consecutive b128 loads
// from a pointer that advances by a constant stride (L2-resident weights).
// ---------------------------------------------------------------------------
__global__ __launch_bounds__(256) void conv_implicit_gemm_wmma(
    const u16t* __restrict__ act, const u16t* __restrict__ wfrag,
    const u16t* __restrict__ zpad, float* __restrict__ out, ConvDesc d)
{
  const int lane = threadIdx.x & 31;
  const int wave = threadIdx.x >> 5;
  const int half = lane >> 4;
  const int lr   = lane & 15;
  const int mbase = blockIdx.x * 128 + wave * 16;
  if (mbase >= d.M) return;                    // wave-uniform exit
  const int nbase  = blockIdx.y * 64;
  const int ntiles = d.Cout >> 4;

  // Decompose this lane's M row into (batch, output coords) -> input base.
  // (Division only here, once per kernel.)
  int m = mbase + lr;
  const bool mok = (m < d.M);
  int rem = mok ? m : 0;
  int bi[4];
  for (int i = d.nd - 1; i >= 0; --i) {
    int o = rem % d.out_dim[i]; rem /= d.out_dim[i];
    bi[i] = o * d.stride[i] - d.pad[i];
  }
  long long nsz = d.Cin;
  for (int i = 0; i < d.nd; ++i) nsz *= d.in_dim[i];
  const u16t* actn = act + (long long)rem * nsz;   // rem == batch index

  const v8f vzero = {0.f,0.f,0.f,0.f,0.f,0.f,0.f,0.f};
  v8f acc0 = vzero, acc1 = vzero, acc2 = vzero, acc3 = vzero;

  const int cin = d.Cin;
  const long long bstep = (long long)ntiles * 512;   // elements per K-step
  const u16t* bptr = wfrag + (((long long)(nbase >> 4)) * 32 + lane) * 16;
  const u16t* zp = zpad + half * 8;

  int kc[4] = {0, 0, 0, 0};                          // tap coords (mixed radix)
  for (int tap = 0; tap < d.taps; ++tap) {
    // Per-tap (loop-invariant over channels): bounds check + spatial offset.
    bool ok = mok;
    long long soff = 0;
    for (int i = 0; i < d.nd; ++i) {
      int x = bi[i] + kc[i];
      ok &= ((unsigned)x < (unsigned)d.in_dim[i]);
      soff = soff * d.in_dim[i] + (x < 0 ? 0 : x);
    }
    // Zero-page redirect: single 64-bit select per tap, loads unconditional.
    const u16t* ap = ok ? (actn + soff * cin + half * 8) : zp;

#pragma unroll 2
    for (int cb = 0; cb < cin; cb += 32) {
      ABFrag a;
      a.q[0] = *(const v4u*)(ap + cb);
      a.q[1] = *(const v4u*)(ap + cb + 16);

      __builtin_prefetch(bptr + bstep, 0, 2);   // next K-step's fragments
      ABFrag b0, b1, b2, b3;
      { const v4u* p = (const v4u*)(bptr);        b0.q[0]=p[0]; b0.q[1]=p[1]; }
      { const v4u* p = (const v4u*)(bptr + 512);  b1.q[0]=p[0]; b1.q[1]=p[1]; }
      { const v4u* p = (const v4u*)(bptr + 1024); b2.q[0]=p[0]; b2.q[1]=p[1]; }
      { const v4u* p = (const v4u*)(bptr + 1536); b3.q[0]=p[0]; b3.q[1]=p[1]; }
      bptr += bstep;

      acc0 = __builtin_amdgcn_wmma_f32_16x16x32_bf16(false, a.v, false, b0.v, (short)0, acc0, false, false);
      acc1 = __builtin_amdgcn_wmma_f32_16x16x32_bf16(false, a.v, false, b1.v, (short)0, acc1, false, false);
      acc2 = __builtin_amdgcn_wmma_f32_16x16x32_bf16(false, a.v, false, b2.v, (short)0, acc2, false, false);
      acc3 = __builtin_amdgcn_wmma_f32_16x16x32_bf16(false, a.v, false, b3.v, (short)0, acc3, false, false);
    }

    // Mixed-radix tap increment: adds/compares only, no div/mod.
    for (int i = d.nd - 1; i >= 0; --i) {
      if (++kc[i] < d.ks[i]) break;
      kc[i] = 0;
    }
  }

  // C/D layout: lane = column N, VGPR r = row M = r + half*8.
#pragma unroll
  for (int r = 0; r < 8; ++r) {
    int mm = mbase + half * 8 + r;
    if (mm < d.M) {
      float* p = out + (long long)mm * d.Cout + nbase + lr;
      p[0]  = acc0[r];
      p[16] = acc1[r];
      p[32] = acc2[r];
      p[48] = acc3[r];
    }
  }
}

// ---------------------------------------------------------------------------
// Weight pre-pack: fp32 (O, I, taps...) -> bf16 WMMA B fragments
// frag[((kt*ntiles + nt)*32 + lane)*16 + j] = W[k = kt*32 + (j<8 ? (lane>>4)*8 + j : 16 + (lane>>4)*8 + j-8)]
//                                              [n = nt*16 + (lane&15)]
// with k = tap*Cin + cin (tap flattened in the weight's trailing-dim order).
// ---------------------------------------------------------------------------
struct WDesc { int Cin, Cout, taps; long long total; };

__global__ void repack_weight_kernel(const float* __restrict__ w,
                                     u16t* __restrict__ frag, WDesc d)
{
  long long idx = (long long)blockIdx.x * blockDim.x + threadIdx.x;
  if (idx >= d.total) return;
  int j = (int)(idx & 15);
  long long t = idx >> 4;
  int lane = (int)(t & 31); t >>= 5;
  int ntiles = d.Cout >> 4;
  int nt = (int)(t % ntiles);
  long long kt = t / ntiles;
  int k = (int)(kt * 32 + (j < 8 ? (lane >> 4) * 8 + j : 16 + (lane >> 4) * 8 + (j - 8)));
  int n = nt * 16 + (lane & 15);
  int tap = k / d.Cin, c = k - tap * d.Cin;
  float v = w[((long long)n * d.Cin + c) * d.taps + tap];
  frag[idx] = f2bf(v);
}

// NCDHWS fp32 -> channels-last bf16 [N,D,H,W,S,C]
__global__ void permute_in_kernel(const float* __restrict__ x, u16t* __restrict__ a,
                                  int N, int C, int d0, int d1, int d2, int d3,
                                  long long total)
{
  long long i = (long long)blockIdx.x * blockDim.x + threadIdx.x;
  if (i >= total) return;
  int c = (int)(i % C); long long t = i / C;
  int s3 = (int)(t % d3); t /= d3;
  int s2 = (int)(t % d2); t /= d2;
  int s1 = (int)(t % d1); t /= d1;
  int s0 = (int)(t % d0); int n = (int)(t / d0);
  long long src = (((((long long)n * C + c) * d0 + s0) * d1 + s1) * d2 + s2) * d3 + s3;
  a[i] = f2bf(x[src]);
}

// Per-channel batch statistics -> fused scale/bias (matches reference _bn).
__global__ __launch_bounds__(256) void bn_stats_kernel(
    const float* __restrict__ y, const float* __restrict__ g,
    const float* __restrict__ b, float* __restrict__ scale,
    float* __restrict__ bias, long long M, int C)
{
  __shared__ float ss[256], sq[256];
  int c = blockIdx.x;
  float s0 = 0.f, s1 = 0.f;
  for (long long i = threadIdx.x; i < M; i += 256) {
    float v = y[i * C + c]; s0 += v; s1 += v * v;
  }
  ss[threadIdx.x] = s0; sq[threadIdx.x] = s1;
  __syncthreads();
  for (int st = 128; st > 0; st >>= 1) {
    if ((int)threadIdx.x < st) {
      ss[threadIdx.x] += ss[threadIdx.x + st];
      sq[threadIdx.x] += sq[threadIdx.x + st];
    }
    __syncthreads();
  }
  if (threadIdx.x == 0) {
    float inv  = 1.f / (float)M;
    float mean = ss[0] * inv;
    float var  = fmaxf(sq[0] * inv - mean * mean, 0.f);
    float sc   = g[c] * rsqrtf(var + 1e-5f);
    scale[c] = sc; bias[c] = b[c] - mean * sc;
  }
}

__global__ void bn_relu_kernel(const float* __restrict__ y,
                               const float* __restrict__ sc,
                               const float* __restrict__ bi,
                               u16t* __restrict__ o, long long total, int C)
{
  long long i = (long long)blockIdx.x * blockDim.x + threadIdx.x;
  if (i >= total) return;
  int c = (int)(i % C);
  float v = fmaf(y[i], sc[c], bi[c]);
  o[i] = f2bf(v > 0.f ? v : 0.f);
}

__global__ void bn_add_relu_kernel(const float* __restrict__ y,
                                   const float* __restrict__ sy,
                                   const float* __restrict__ by,
                                   const float* __restrict__ z,
                                   const float* __restrict__ sz,
                                   const float* __restrict__ bz,
                                   u16t* __restrict__ o, long long total, int C)
{
  long long i = (long long)blockIdx.x * blockDim.x + threadIdx.x;
  if (i >= total) return;
  int c = (int)(i % C);
  float v = fmaf(y[i], sy[c], by[c]) + fmaf(z[i], sz[c], bz[c]);
  o[i] = f2bf(v > 0.f ? v : 0.f);
}

__global__ void pool_kernel(const u16t* __restrict__ act, float* __restrict__ out,
                            int spatial, int C, int N)
{
  int idx = blockIdx.x * blockDim.x + threadIdx.x;
  if (idx >= N * C) return;
  int n = idx / C, c = idx % C;
  float s = 0.f;
  for (int i = 0; i < spatial; ++i)
    s += bf2f(act[((long long)n * spatial + i) * C + c]);
  out[idx] = s / (float)spatial;
}

__global__ __launch_bounds__(256) void fc_kernel(
    const float* __restrict__ x, const float* __restrict__ w,
    const float* __restrict__ b, float* __restrict__ out, int K, int O)
{
  __shared__ float red[256];
  int n = blockIdx.x / O, o = blockIdx.x % O;
  float s = 0.f;
  for (int k = threadIdx.x; k < K; k += 256)
    s += x[n * K + k] * w[(long long)o * K + k];
  red[threadIdx.x] = s;
  __syncthreads();
  for (int st = 128; st > 0; st >>= 1) {
    if ((int)threadIdx.x < st) red[threadIdx.x] += red[threadIdx.x + st];
    __syncthreads();
  }
  if (threadIdx.x == 0) out[blockIdx.x] = red[0] + b[o];
}

// ---------------------------------------------------------------------------
// Host orchestration
// ---------------------------------------------------------------------------
namespace {

struct Alloc {
  char* base; size_t off, cap;
  void* get(size_t bytes) {
    size_t a = (off + 255) & ~(size_t)255;
    off = a + bytes;
    return (void*)(base + a);
  }
};

struct Shape { int nd; int sp[4]; int C; };

long long mcountS(int N, const Shape& s) {
  long long m = N;
  for (int i = 0; i < s.nd; ++i) m *= s.sp[i];
  return m;
}

ConvDesc mk(int N, const Shape& in, int Cout, const int* ks, int stride,
            const int* pad, Shape& out) {
  ConvDesc d{};
  d.nd = in.nd; d.N = N; d.Cin = in.C; d.Cout = Cout;
  int taps = 1; long long M = N;
  out.nd = in.nd; out.C = Cout;
  for (int i = 0; i < in.nd; ++i) {
    d.in_dim[i] = in.sp[i]; d.ks[i] = ks[i]; d.stride[i] = stride; d.pad[i] = pad[i];
    int o = (in.sp[i] + 2 * pad[i] - ks[i]) / stride + 1;
    d.out_dim[i] = o; out.sp[i] = o; taps *= ks[i]; M *= o;
  }
  d.taps = taps; d.K = in.C * taps; d.M = (int)M;
  return d;
}

const u16t* pack_w(hipStream_t st, Alloc& al, const float* w,
                   int Cout, int Cin, int taps) {
  long long total = (long long)Cin * taps * Cout;
  u16t* frag = (u16t*)al.get((size_t)total * 2);
  WDesc wd{Cin, Cout, taps, total};
  repack_weight_kernel<<<dim3((unsigned)((total + 255) / 256)), dim3(256), 0, st>>>(
      w, frag, wd);
  return frag;
}

void run_conv(hipStream_t st, const u16t* act, const u16t* wf, const u16t* zpad,
              float* out, const ConvDesc& d) {
  dim3 grid((unsigned)((d.M + 127) / 128), (unsigned)(d.Cout / 64));
  conv_implicit_gemm_wmma<<<grid, dim3(256), 0, st>>>(act, wf, zpad, out, d);
}

float* conv_bn(hipStream_t st, Alloc& al, int N, const Shape& in, const u16t* act,
               const u16t* zpad, const float* w, const float* g, const float* b,
               int Cout, const int* ks, int stride, const int* pad,
               Shape& outShape, float** scale, float** bias) {
  ConvDesc d = mk(N, in, Cout, ks, stride, pad, outShape);
  const u16t* wf = pack_w(st, al, w, Cout, in.C, d.taps);
  long long M = mcountS(N, outShape);
  float* Y = (float*)al.get((size_t)(M * Cout) * 4);
  run_conv(st, act, wf, zpad, Y, d);
  float* sc = (float*)al.get((size_t)Cout * 4);
  float* bi = (float*)al.get((size_t)Cout * 4);
  bn_stats_kernel<<<dim3((unsigned)Cout), dim3(256), 0, st>>>(Y, g, b, sc, bi, M, Cout);
  *scale = sc; *bias = bi;
  return Y;
}

const u16t* run_block(hipStream_t st, Alloc& al, int N, Shape& s,
                      const u16t* X, const u16t* zpad, int planes,
                      void* const* din, int stride) {
  const float* w1 = (const float*)din[0];
  const float* g1 = (const float*)din[1];
  const float* b1 = (const float*)din[2];
  const float* w2 = (const float*)din[3];
  const float* g2 = (const float*)din[4];
  const float* b2 = (const float*)din[5];
  const float* w3 = (const float*)din[6];
  const float* g3 = (const float*)din[7];
  const float* b3 = (const float*)din[8];
  const float* wd = (const float*)din[9];
  const float* gd = (const float*)din[10];
  const float* bd = (const float*)din[11];
  const int ones[4] = {1,1,1,1}, zer[4] = {0,0,0,0};
  const int thr[4]  = {3,3,3,3}, pd1[4] = {1,1,1,1};
  const int Cexp = planes * 4;

  // conv1 (1^nd) -> bn -> relu
  Shape s1; float *sc1, *bi1;
  float* Y1 = conv_bn(st, al, N, s, X, zpad, w1, g1, b1, planes, ones, 1, zer,
                      s1, &sc1, &bi1);
  long long M1 = mcountS(N, s1);
  u16t* A1 = (u16t*)al.get((size_t)(M1 * planes) * 2);
  {
    long long t = M1 * planes;
    bn_relu_kernel<<<dim3((unsigned)((t + 255) / 256)), dim3(256), 0, st>>>(
        Y1, sc1, bi1, A1, t, planes);
  }

  // conv2 (3^nd, stride, pad 1) -> bn -> relu
  Shape s2; float *sc2, *bi2;
  float* Y2 = conv_bn(st, al, N, s1, A1, zpad, w2, g2, b2, planes, thr, stride,
                      pd1, s2, &sc2, &bi2);
  long long M2 = mcountS(N, s2);
  u16t* A2 = (u16t*)al.get((size_t)(M2 * planes) * 2);
  {
    long long t = M2 * planes;
    bn_relu_kernel<<<dim3((unsigned)((t + 255) / 256)), dim3(256), 0, st>>>(
        Y2, sc2, bi2, A2, t, planes);
  }

  // conv3 (1^nd, expand) -> bn
  Shape s3; float *sc3, *bi3;
  float* Y3 = conv_bn(st, al, N, s2, A2, zpad, w3, g3, b3, Cexp, ones, 1, zer,
                      s3, &sc3, &bi3);

  // downsample path: conv (1^nd, stride) -> bn
  Shape sd; float *scd, *bid;
  float* Yd = conv_bn(st, al, N, s, X, zpad, wd, gd, bd, Cexp, ones, stride, zer,
                      sd, &scd, &bid);

  // relu(bn(Y3) + bn(Yd)) -> bf16
  long long t3 = mcountS(N, s3) * Cexp;
  u16t* OUT = (u16t*)al.get((size_t)t3 * 2);
  bn_add_relu_kernel<<<dim3((unsigned)((t3 + 255) / 256)), dim3(256), 0, st>>>(
      Y3, sc3, bi3, Yd, scd, bid, OUT, t3, Cexp);
  s = s3;
  return OUT;
}

} // namespace

extern "C" void kernel_launch(void* const* d_in, const int* in_sizes, int n_in,
                              void* d_out, int out_size, void* d_ws, size_t ws_size,
                              hipStream_t stream) {
  (void)in_sizes; (void)n_in; (void)out_size;
  Alloc al{(char*)d_ws, 0, ws_size};
  const int N = 2;

  // Zero pad page for out-of-bounds conv taps (ws is poisoned; re-zero per call).
  u16t* zpad = (u16t*)al.get(ZPAD_ELEMS * 2);
  zero_fill_kernel<<<dim3(ZPAD_ELEMS / 256), dim3(256), 0, stream>>>(zpad, ZPAD_ELEMS);

  // stage 0: x (2,64,8,28,28,4) fp32 NCDHWS -> channels-last bf16
  Shape s; s.nd = 4; s.sp[0] = 8; s.sp[1] = 28; s.sp[2] = 28; s.sp[3] = 4; s.C = 64;
  long long M0 = mcountS(N, s);
  long long tot0 = M0 * s.C;
  u16t* A0 = (u16t*)al.get((size_t)tot0 * 2);
  permute_in_kernel<<<dim3((unsigned)((tot0 + 255) / 256)), dim3(256), 0, stream>>>(
      (const float*)d_in[0], A0, N, s.C, 8, 28, 28, 4, tot0);

  const u16t* act = A0;
  act = run_block(stream, al, N, s, act, zpad, 64,  d_in + 1,  1);   // l1
  act = run_block(stream, al, N, s, act, zpad, 128, d_in + 13, 2);   // l2

  // dimension-reduction conv (512->512, kernel (1,1,1,2)) -> bn -> relu, squeeze S
  {
    const int ks[4] = {1,1,1,2}, pad0[4] = {0,0,0,0};
    Shape so; float *sc, *bi;
    float* Y = conv_bn(stream, al, N, s, act, zpad, (const float*)d_in[25],
                       (const float*)d_in[26], (const float*)d_in[27],
                       512, ks, 1, pad0, so, &sc, &bi);
    long long M = mcountS(N, so);
    long long t = M * 512;
    u16t* A = (u16t*)al.get((size_t)t * 2);
    bn_relu_kernel<<<dim3((unsigned)((t + 255) / 256)), dim3(256), 0, stream>>>(
        Y, sc, bi, A, t, 512);
    s = so;
    s.nd = 3;   // squeeze trailing S==1 dim: channels-last memory unchanged
    act = A;
  }

  act = run_block(stream, al, N, s, act, zpad, 256, d_in + 28, 2);   // l3
  act = run_block(stream, al, N, s, act, zpad, 512, d_in + 40, 2);   // l4

  // global average pool over spatial -> (2, 2048), then FC -> (2, 400)
  int spatial = 1;
  for (int i = 0; i < s.nd; ++i) spatial *= s.sp[i];
  const int C = s.C;   // 2048
  float* pooled = (float*)al.get((size_t)(N * C) * 4);
  pool_kernel<<<dim3((unsigned)((N * C + 255) / 256)), dim3(256), 0, stream>>>(
      act, pooled, spatial, C, N);
  fc_kernel<<<dim3((unsigned)(N * 400)), dim3(256), 0, stream>>>(
      pooled, (const float*)d_in[52], (const float*)d_in[53], (float*)d_out, C, 400);
}
